// FusedExpertsNetwork_47107201303190
// MI455X (gfx1250) — matrix operand order repaired and checked by reading, not compiled, not run on hardware
//
#include <hip/hip_runtime.h>
#include <hip/hip_bf16.h>

// Problem constants (match reference)
constexpr int Bb = 4, Ee = 8, Cc = 512, Mm = 1024, Hh = 4096;
constexpr int Tt = Bb * Cc; // 2048 tokens per expert

typedef __attribute__((ext_vector_type(16))) __bf16 v16bf;
typedef __attribute__((ext_vector_type(8)))  __bf16 v8bf;
typedef __attribute__((ext_vector_type(4)))  __bf16 v4bf;
typedef __attribute__((ext_vector_type(8)))  float  v8f;
typedef __attribute__((ext_vector_type(4)))  float  v4f;

typedef __attribute__((ext_vector_type(4))) unsigned int u32x4;
typedef __attribute__((ext_vector_type(8))) int i32x8;
typedef __attribute__((ext_vector_type(4))) int i32x4;

union V16 { v16bf v; v8bf h[2]; };

constexpr int TILE = 128;  // macro tile (rows=tokens, cols=N)
constexpr int KC   = 32;   // k-step (matches 16x16x32 bf16 WMMA)
constexpr int LDST = 40;   // padded LDS row stride (halves): 64B data + 16B pad

// ---- CDNA5 async / TDM availability (guarded; fall back to sync staging) ----
#if defined(__HIP_DEVICE_COMPILE__) && __has_builtin(__builtin_amdgcn_global_load_async_to_lds_b128)
#define USE_ASYNC 1
#else
#define USE_ASYNC 0
#endif
#if defined(__HIP_DEVICE_COMPILE__) && __has_builtin(__builtin_amdgcn_tensor_load_to_lds)
#define USE_TDM 1
#else
#define USE_TDM 0
#endif

#define AS1 __attribute__((address_space(1)))
#define AS3 __attribute__((address_space(3)))

// Builtin parameter types (from hipcc diagnostic): global side is
// addrspace(1) int4*, LDS side addrspace(3) int4*.
typedef AS1 i32x4* gptr_b128;
typedef AS3 i32x4* lptr_b128;

__device__ __forceinline__ gptr_b128 as_global_b128(const void* p) {
  return (gptr_b128)(unsigned long long)p;
}
__device__ __forceinline__ lptr_b128 as_lds_b128(const void* p) {
  // LDS aperture occupies only the high 32 bits of a flat address (ISA 10.2),
  // so truncating a flat shared pointer yields the LDS byte offset.
  return (lptr_b128)(unsigned long long)p;
}
__device__ __forceinline__ unsigned lds_offset_of(const void* p) {
  return (unsigned)(unsigned long long)p;
}

// ---------------------------------------------------------------------------
// Conversion kernels (fp32 -> bf16, plus layout changes)
// ---------------------------------------------------------------------------

// xe[e][t][m] = x[b][e][c][m], t = b*C + c
__global__ __launch_bounds__(256) void convert_x_kernel(
    const float* __restrict__ x, __bf16* __restrict__ xb) {
  size_t idx4 = ((size_t)blockIdx.x * blockDim.x + threadIdx.x) * 4;
  size_t total = (size_t)Ee * Tt * Mm;
  if (idx4 >= total) return;
  int m = (int)(idx4 % Mm);
  size_t rest = idx4 / Mm;
  int t = (int)(rest % Tt);
  int e = (int)(rest / Tt);
  int b = t / Cc, c = t % Cc;
  const float* src = x + ((((size_t)b * Ee + e) * Cc + c) * Mm + m);
  v4f f = *(const v4f*)src;
  *(v4bf*)(xb + idx4) = __builtin_convertvector(f, v4bf);
}

__global__ __launch_bounds__(256) void convert_lin_kernel(
    const float* __restrict__ src, __bf16* __restrict__ dst, size_t total) {
  size_t idx4 = ((size_t)blockIdx.x * blockDim.x + threadIdx.x) * 4;
  if (idx4 >= total) return;
  v4f f = *(const v4f*)(src + idx4);
  *(v4bf*)(dst + idx4) = __builtin_convertvector(f, v4bf);
}

// w2t[e][m][h] = bf16(w2[e][h][m])  — 32x32 LDS-tiled transpose
__global__ __launch_bounds__(256) void transpose_w2_kernel(
    const float* __restrict__ w2, __bf16* __restrict__ w2t) {
  __shared__ float tile[32][33];
  const int e  = blockIdx.z;
  const int m0 = blockIdx.x * 32;
  const int h0 = blockIdx.y * 32;
  const float* src = w2 + (size_t)e * Hh * Mm;
  __bf16*      dst = w2t + (size_t)e * Mm * Hh;
  const int tx = threadIdx.x, ty = threadIdx.y; // 32 x 8
#pragma unroll
  for (int k = 0; k < 4; ++k)
    tile[ty + k * 8][tx] = src[(size_t)(h0 + ty + k * 8) * Mm + (m0 + tx)];
  __syncthreads();
#pragma unroll
  for (int k = 0; k < 4; ++k)
    dst[(size_t)(m0 + ty + k * 8) * Hh + (h0 + tx)] = (__bf16)tile[tx][ty + k * 8];
}

// ---------------------------------------------------------------------------
// WMMA grouped GEMM: D[t][n] = sum_k A[t][k] * W[n][k]  (NT, K contiguous)
// ---------------------------------------------------------------------------
template <int KDIM, bool IS_FC1>
__global__ __launch_bounds__(256, 2) void wmma_gemm_kernel(
    const __bf16* __restrict__ Aall, const __bf16* __restrict__ Wall,
    const float* __restrict__ biasAll, __bf16* __restrict__ hOut,
    float* __restrict__ oOut) {
  constexpr int N = IS_FC1 ? Hh : Mm;
  const int e  = blockIdx.z;
  const int t0 = blockIdx.y * TILE;
  const int n0 = blockIdx.x * TILE;

  const __bf16* A = Aall + (size_t)e * Tt * KDIM + (size_t)t0 * KDIM;
  const __bf16* W = Wall + (size_t)e * (size_t)N * KDIM + (size_t)n0 * KDIM;
  const float* bias = biasAll + (size_t)e * N + n0;

  __shared__ __attribute__((aligned(16))) __bf16 sA[TILE * LDST];
  __shared__ __attribute__((aligned(16))) __bf16 sB[TILE * LDST];

  const int tid  = threadIdx.x;
  const int lane = tid & 31;
  const int wave = tid >> 5;  // 0..7
  const int wm   = wave & 3;  // 4 m-bands of 32 rows
  const int wn   = wave >> 2; // 2 n-bands of 64 cols
  const int hs   = lane >> 4; // half-wave select (ISA 16-bit operand layout)
  const int lr   = lane & 15;

  v8f acc[2][4];
#pragma unroll
  for (int i = 0; i < 2; ++i)
#pragma unroll
    for (int j = 0; j < 4; ++j) {
      v8f z = {0.f, 0.f, 0.f, 0.f, 0.f, 0.f, 0.f, 0.f};
      acc[i][j] = z;
    }

  // cooperative tile loader mapping: 4 threads x 8 halves per row, 64 rows/pass
  const int lrow = tid >> 2;
  const int lcol = (tid & 3) * 8;

  for (int k0 = 0; k0 < KDIM; k0 += KC) {
    // ---- B tile: 128 N-rows x 32 K (tensor DMA if available) ----
#if USE_TDM
    if (tid < 32) {
      // Tensor DMA Descriptor (ISA cdna5 §8): 2D tensor KDIM x N, tile 32x128,
      // bf16 (data_size=1 -> 2B), LDS padding 16 DWORDs data + 4 DWORDs pad
      // per row => LDS row stride 40 halves (== LDST).
      unsigned long long ga = (unsigned long long)(const void*)(W + k0);
      unsigned lds = lds_offset_of(sB);
      u32x4 g0 = {1u /*count=1, user desc*/, lds,
                  (unsigned)(ga & 0xffffffffu),
                  (unsigned)((ga >> 32) & 0x01ffffffu) | (2u << 30) /*type=2*/};
      i32x8 g1 = {
          (int)((1u << 16) /*data_size=2B*/ | (1u << 20) /*pad_enable*/ |
                (3u << 22) /*pad_interval: 16 DW*/ | (3u << 25) /*pad: 4 DW*/),
          (int)(((unsigned)KDIM & 0xffffu) << 16),           // tensor_dim0 lo
          (int)((((unsigned)KDIM >> 16) & 0xffffu) |
                (((unsigned)N & 0xffffu) << 16)),            // dim0 hi | dim1 lo
          (int)((((unsigned)N >> 16) & 0xffffu) | (32u << 16)), // dim1 hi | tile_dim0
          (int)(128u),                                       // tile_dim1 | tile_dim2=0
          (int)(unsigned)KDIM,                               // tensor_dim0_stride lo
          0, 0};
      i32x4 gz = {0, 0, 0, 0};
#if __clang_major__ >= 23
      i32x8 gz8 = {0, 0, 0, 0, 0, 0, 0, 0};
      __builtin_amdgcn_tensor_load_to_lds(g0, g1, gz, gz, gz8, 0);
#else
      __builtin_amdgcn_tensor_load_to_lds(g0, g1, gz, gz, 0);
#endif
    }
#else
#pragma unroll
    for (int p = 0; p < 2; ++p) {
      const int r = lrow + p * 64;
#if USE_ASYNC
      __builtin_amdgcn_global_load_async_to_lds_b128(
          as_global_b128(W + (size_t)r * KDIM + k0 + lcol),
          as_lds_b128(sB + r * LDST + lcol), 0, 0);
#else
      v8bf vb = *(const v8bf*)(W + (size_t)r * KDIM + k0 + lcol);
      *(v8bf*)(sB + r * LDST + lcol) = vb;
#endif
    }
#endif

    // ---- A tile: 128 token-rows x 32 K ----
#pragma unroll
    for (int p = 0; p < 2; ++p) {
      const int r = lrow + p * 64;
#if USE_ASYNC
      __builtin_amdgcn_global_load_async_to_lds_b128(
          as_global_b128(A + (size_t)r * KDIM + k0 + lcol),
          as_lds_b128(sA + r * LDST + lcol), 0, 0);
#else
      v8bf va = *(const v8bf*)(A + (size_t)r * KDIM + k0 + lcol);
      *(v8bf*)(sA + r * LDST + lcol) = va;
#endif
    }
#if !USE_ASYNC
    if (k0 + KC < KDIM) {
      __builtin_prefetch(A + (size_t)lrow * KDIM + k0 + KC + lcol, 0, 1);
      __builtin_prefetch(W + (size_t)lrow * KDIM + k0 + KC + lcol, 0, 1);
    }
#endif

    // ---- waits + barrier ----
#if USE_ASYNC
#if __has_builtin(__builtin_amdgcn_s_wait_asynccnt)
    __builtin_amdgcn_s_wait_asynccnt(0);
#else
    asm volatile("s_wait_asynccnt 0x0" ::: "memory");
#endif
#endif
#if USE_TDM
    __builtin_amdgcn_s_wait_tensorcnt(0); // no-op for waves with TENSORcnt==0
#endif
    __syncthreads();

    // A fragments: 16x32 bf16, lanes 0-15 hold K 0..7 & 16..23 of row M=lane,
    // lanes 16-31 hold K 8..15 & 24..31 of row M=lane-16 (ISA 7.12.2).
    V16 afrag[2];
#pragma unroll
    for (int i = 0; i < 2; ++i) {
      const __bf16* p = sA + (wm * 32 + i * 16 + lr) * LDST + hs * 8;
      afrag[i].h[0] = *(const v8bf*)(p);
      afrag[i].h[1] = *(const v8bf*)(p + 16);
    }
    // B fragments: 32x16; lane holds one N-column = one W row: K 0..7 & 16..23 (lanes 0-15), K 8..15 & 24..31 (lanes 16-31).
    V16 bfrag[4];
#pragma unroll
    for (int j = 0; j < 4; ++j) {
      const __bf16* p = sB + (wn * 64 + j * 16 + lr) * LDST + hs * 8;
      bfrag[j].h[0] = *(const v8bf*)(p);
      bfrag[j].h[1] = *(const v8bf*)(p + 16);
    }

#pragma unroll
    for (int i = 0; i < 2; ++i)
#pragma unroll
      for (int j = 0; j < 4; ++j)
        acc[i][j] = __builtin_amdgcn_wmma_f32_16x16x32_bf16(
            false, afrag[i].v, false, bfrag[j].v, (short)0, acc[i][j], false,
            false);

    __syncthreads();
  }

  // Epilogue. C/D layout: VGPR r, lanes 0-15 -> M=r, N=lane; lanes 16-31 -> M=8+r.
#pragma unroll
  for (int i = 0; i < 2; ++i) {
#pragma unroll
    for (int j = 0; j < 4; ++j) {
      const int nloc = wn * 64 + j * 16 + lr;
      const int col  = n0 + nloc;
      const float bv = bias[nloc];
#pragma unroll
      for (int r = 0; r < 8; ++r) {
        const int t = t0 + wm * 32 + i * 16 + hs * 8 + r;
        float val = acc[i][j][r] + bv;
        if (IS_FC1) {
          val = fmaxf(val, 0.0f);
          hOut[(size_t)e * Tt * Hh + (size_t)t * Hh + col] = (__bf16)val;
        } else {
          const int b = t / Cc, c = t % Cc;
          oOut[(((size_t)b * Ee + e) * Cc + c) * Mm + col] = val;
        }
      }
    }
  }
}

// ---------------------------------------------------------------------------
extern "C" void kernel_launch(void* const* d_in, const int* in_sizes, int n_in,
                              void* d_out, int out_size, void* d_ws,
                              size_t ws_size, hipStream_t stream) {
  (void)in_sizes; (void)n_in; (void)out_size; (void)ws_size;
  const float* x  = (const float*)d_in[0];
  const float* w1 = (const float*)d_in[1];
  const float* b1 = (const float*)d_in[2];
  const float* w2 = (const float*)d_in[3];
  const float* b2 = (const float*)d_in[4];
  float* out = (float*)d_out;

  // workspace layout (bf16): xe[E,T,M] | w1[E,H,M] | w2t[E,M,H] | h[E,T,H]
  char* ws = (char*)d_ws;
  size_t off = 0;
  __bf16* xb  = (__bf16*)(ws + off); off += (size_t)Ee * Tt * Mm * 2;
  __bf16* w1b = (__bf16*)(ws + off); off += (size_t)Ee * Hh * Mm * 2;
  __bf16* w2t = (__bf16*)(ws + off); off += (size_t)Ee * Mm * Hh * 2;
  __bf16* hbf = (__bf16*)(ws + off);

  {
    size_t total = (size_t)Ee * Tt * Mm;
    convert_x_kernel<<<dim3((unsigned)((total / 4 + 255) / 256)), 256, 0,
                       stream>>>(x, xb);
  }
  {
    size_t total = (size_t)Ee * Hh * Mm;
    convert_lin_kernel<<<dim3((unsigned)((total / 4 + 255) / 256)), 256, 0,
                         stream>>>(w1, w1b, total);
  }
  transpose_w2_kernel<<<dim3(Mm / 32, Hh / 32, Ee), dim3(32, 8), 0, stream>>>(
      w2, w2t);

  // GEMM1: h = relu(xe @ W1^T + b1)  -> bf16 [E,T,H]
  wmma_gemm_kernel<Mm, true>
      <<<dim3(Hh / TILE, Tt / TILE, Ee), 256, 0, stream>>>(xb, w1b, b1, hbf,
                                                           nullptr);
  // GEMM2: out = h @ W2 + b2 -> fp32 [B,E,C,M] (scattered in epilogue)
  wmma_gemm_kernel<Hh, false>
      <<<dim3(Mm / TILE, Tt / TILE, Ee), 256, 0, stream>>>(hbf, w2t, b2,
                                                           nullptr, out);
}
